// Net_412316860624
// MI455X (gfx1250) — matrix operand order repaired and checked by reading, not compiled, not run on hardware
//
#include <hip/hip_runtime.h>
#include <hip/hip_bf16.h>

// ---------------------------------------------------------------------------
// CDNA5 / gfx1250 implementation of the 20-layer conv + k-max-sort network.
// Layout convention for activations: [C][W][H], H contiguous.
// - convs + FC on V_WMMA_F32_16X16X4_F32 (f16 WMMA fallback)
// - per-column descending bitonic sort with 64KB LDS-resident segments
// - LDS staging via the Tensor Data Mover (tensor_load_to_lds + TENSORcnt);
//   this toolchain's builtin has the 6-arg (clang-23) arity
// ---------------------------------------------------------------------------

typedef __attribute__((ext_vector_type(2)))  float        v2f;
typedef __attribute__((ext_vector_type(8)))  float        v8f;
typedef __attribute__((ext_vector_type(16))) _Float16     v16h;
typedef __attribute__((ext_vector_type(4)))  unsigned int u32x4;
typedef __attribute__((ext_vector_type(4)))  int          i32x4;
typedef __attribute__((ext_vector_type(8)))  int          i32x8;

#if defined(__has_builtin)
#  if __has_builtin(__builtin_amdgcn_wmma_f32_16x16x4_f32)
#    define HAS_WMMA_F32F32 1
#  endif
#endif
#ifndef HAS_WMMA_F32F32
#  define HAS_WMMA_F32F32 0
#endif

// Flip to 0 to disable the Tensor Data Mover staging path.
#define ATTEMPT_TDM 1
#if ATTEMPT_TDM && defined(__has_builtin)
#  if __has_builtin(__builtin_amdgcn_tensor_load_to_lds) && \
      __has_builtin(__builtin_amdgcn_s_wait_tensorcnt)
#    define HAS_TDM 1
#  endif
#endif
#ifndef HAS_TDM
#  define HAS_TDM 0
#endif

#define SEGSZ   16384   // floats per LDS-resident sort segment (64 KB)
#define STHREADS 1024

// ---------------------------------------------------------------------------
// Generic 1-D conv (along H) as WMMA GEMM; one wave per (w, 16-position tile).
// M = positions, N = out channels (padded to 16), K = Cin*Ksz.
// ---------------------------------------------------------------------------
__global__ __launch_bounds__(32)
void conv_wmma_kernel(const float* __restrict__ in, const float* __restrict__ wgt,
                      const float* __restrict__ bias, float* __restrict__ out,
                      int Cin, int Cout, int W, int Hin, int Hout, int Ksz, int pad)
{
    const int lane  = threadIdx.x;
    const int m     = lane & 15;
    const int half  = lane >> 4;
    const int tiles = (Hout + 15) >> 4;
    const int w     = blockIdx.x / tiles;
    const int h0    = (blockIdx.x % tiles) << 4;
    const int Ktot  = Cin * Ksz;

    {   // CDNA5 speculative prefetch of the next tile's input span.
        int hp = h0 + 16; if (hp >= Hin) hp = 0;
        __builtin_prefetch(in + (size_t)w * Hin + hp, 0, 0);
    }

    v8f acc = {};

#if HAS_WMMA_F32F32
    for (int kc = 0; kc < Ktot; kc += 4) {
        v2f av, bv;
#pragma unroll
        for (int v = 0; v < 2; ++v) {
            const int kl = kc + 2 * half + v;
            float a = 0.f, b = 0.f;
            if (kl < Ktot) {
                const int ci   = kl / Ksz;
                const int kk   = kl - ci * Ksz;
                const int hsrc = h0 + m + kk - pad;
                if (hsrc >= 0 && hsrc < Hin)
                    a = in[((size_t)ci * W + w) * Hin + hsrc];
                if (m < Cout)
                    b = wgt[((size_t)m * Cin + ci) * Ksz + kk];
            }
            av[v] = a; bv[v] = b;
        }
        acc = __builtin_amdgcn_wmma_f32_16x16x4_f32(
                  false, av, false, bv, (short)0, acc, false, false);
    }
#else
    for (int kc = 0; kc < Ktot; kc += 32) {
        v16h av, bv;
#pragma unroll
        for (int v = 0; v < 8; ++v) {
#pragma unroll
            for (int p = 0; p < 2; ++p) {
                const int kla = kc + 16 * (v >> 2) + 8 * half + 2 * (v & 3) + p;
                float a = 0.f;
                if (kla < Ktot) {
                    const int ci   = kla / Ksz;
                    const int kk   = kla - ci * Ksz;
                    const int hsrc = h0 + m + kk - pad;
                    if (hsrc >= 0 && hsrc < Hin)
                        a = in[((size_t)ci * W + w) * Hin + hsrc];
                }
                const int klb = kc + 16 * (v >> 2) + 8 * half + 2 * (v & 3) + p;
                float b = 0.f;
                if (klb < Ktot && m < Cout) {
                    const int ci = klb / Ksz;
                    const int kk = klb - ci * Ksz;
                    b = wgt[((size_t)m * Cin + ci) * Ksz + kk];
                }
                av[2 * v + p] = (_Float16)a;
                bv[2 * v + p] = (_Float16)b;
            }
        }
        acc = __builtin_amdgcn_wmma_f32_16x16x32_f16(
                  false, av, false, bv, (short)0, acc, false, false);
    }
#endif

    const float bval = (m < Cout) ? bias[m] : 0.f;
#pragma unroll
    for (int r = 0; r < 8; ++r) {
        const int mm = r + 8 * half;
        const int p  = h0 + mm;
        if (m < Cout && p < Hout)
            out[((size_t)m * W + w) * Hout + p] = acc[r] + bval;
    }
}

// ---------------------------------------------------------------------------
// fold(): out[c][w2][j*H + h] = in[c][2w2][h] + in[c][2w2+1][h],  j in [0,C)
// ---------------------------------------------------------------------------
__global__ void fold_kernel(const float* __restrict__ in, float* __restrict__ out,
                            int C, int W, int H)
{
    const int W2 = W >> 1;
    const size_t total = (size_t)C * W2 * (size_t)C * H;
    size_t idx = (size_t)blockIdx.x * blockDim.x + threadIdx.x;
    if (idx >= total) return;
    const int h  = (int)(idx % (size_t)H);
    size_t    t  = idx / (size_t)H;
    t /= (size_t)C;                    // drop replication index j
    const int w2 = (int)(t % (size_t)W2);
    const int c  = (int)(t / (size_t)W2);
    out[idx] = in[((size_t)c * W + 2 * w2) * H + h]
             + in[((size_t)c * W + 2 * w2 + 1) * H + h];
}

// ---------------------------------------------------------------------------
// Sort pipeline.  [cols][n_p] buffer padded with -inf, descending bitonic.
// ---------------------------------------------------------------------------
__global__ void sort_pad_kernel(const float* __restrict__ in, float* __restrict__ buf,
                                int n_p, int cols, int H)
{
    const size_t total = (size_t)cols * n_p;
    size_t idx = (size_t)blockIdx.x * blockDim.x + threadIdx.x;
    if (idx >= total) return;
    const unsigned i   = (unsigned)(idx & (size_t)(n_p - 1));
    const size_t   col = idx / (size_t)n_p;
    buf[idx] = (i < (unsigned)H) ? in[col * (size_t)H + i] : -__builtin_inff();
}

// global pass: one compare-exchange stage (for j >= SEGSZ, or every stage when n_p < SEGSZ)
__global__ void bitonic_step_kernel(float* __restrict__ buf, int n_p, int cols,
                                    unsigned j, unsigned k)
{
    const size_t total = (size_t)cols * n_p;
    size_t tid = (size_t)blockIdx.x * blockDim.x + threadIdx.x;
    if (tid >= total) return;
    const unsigned i   = (unsigned)(tid & (size_t)(n_p - 1));
    const unsigned ixj = i ^ j;
    if (ixj <= i) return;
    float* d = buf + (tid - i);
    const float a = d[i];
    const float b = d[ixj];
    const bool  up   = ((i & k) == 0);
    const bool  swap = up ? (a < b) : (a > b);   // overall descending
    if (swap) { d[i] = b; d[ixj] = a; }
}

// LDS staging helper: Tensor Data Mover bulk load if available.
__device__ __forceinline__ void stage_segment_to_lds(float* s, const float* g)
{
#if HAS_TDM
    if (threadIdx.x < 32) {
        const unsigned long long ga = (unsigned long long)(uintptr_t)g;
        u32x4 g0;
        g0[0] = 1u;                                          // count=1 (valid user D#)
        g0[1] = 0u;                                          // lds_addr = 0 (single __shared__)
        g0[2] = (unsigned)(ga & 0xFFFFFFFFu);                // global_addr lo
        g0[3] = (unsigned)((ga >> 32) & 0x01FFFFFFu) | (2u << 30);  // addr hi | type=2
        i32x8 g1;
        g1[0] = 0x20000;                                     // data_size = 2 (4 bytes)
        g1[1] = (int)((SEGSZ & 0xFFFF) << 16);               // tensor_dim0[15:0]
        g1[2] = (int)((SEGSZ >> 16) | (1 << 16));            // tensor_dim0[31:16] | tensor_dim1=1
        g1[3] = (int)((SEGSZ & 0xFFFF) << 16);               // tile_dim0 = SEGSZ
        g1[4] = 1;                                           // tile_dim1 = 1
        g1[5] = SEGSZ;                                       // tensor_dim0_stride
        g1[6] = 0; g1[7] = 0;
        i32x4 gz4 = {0, 0, 0, 0};
        i32x8 gz8 = {0, 0, 0, 0, 0, 0, 0, 0};
        // clang-23 6-arg form: (v4u, v8i, v4i, v4i, v8i, cpol)
        __builtin_amdgcn_tensor_load_to_lds(g0, g1, gz4, gz4, gz8, 0);
        __builtin_amdgcn_s_wait_tensorcnt(0);
    }
    __syncthreads();
#else
    for (int t = threadIdx.x; t < SEGSZ / 4; t += STHREADS)
        ((float4*)s)[t] = ((const float4*)g)[t];
    __syncthreads();
#endif
}

// LDS-resident bitonic kernels.  FULL=true: all stages k=2..SEGSZ for the
// segment.  FULL=false: merge stages j=SEGSZ/2..1 for a given k (>SEGSZ).
// Directions use the global index within the column (base + i).
template <bool FULL>
__global__ __launch_bounds__(STHREADS)
void bitonic_local_kernel(float* __restrict__ buf, int n_p, unsigned kArg)
{
    __shared__ float s[SEGSZ];
    const int segs_per_col = n_p / SEGSZ;
    const int col = blockIdx.x / segs_per_col;
    const int seg = blockIdx.x % segs_per_col;
    float* g = buf + (size_t)col * n_p + (size_t)seg * SEGSZ;
    const unsigned base = (unsigned)seg * SEGSZ;   // index of segment within column

    stage_segment_to_lds(s, g);

    const unsigned kLo = FULL ? 2u : kArg;
    const unsigned kHi = FULL ? (unsigned)SEGSZ : kArg;
    for (unsigned k = kLo; k <= kHi; k <<= 1) {
        const unsigned jStart = FULL ? (k >> 1) : (SEGSZ >> 1);
        for (unsigned j = jStart; j > 0; j >>= 1) {
#pragma unroll
            for (int it = 0; it < SEGSZ / (2 * STHREADS); ++it) {
                const unsigned t = threadIdx.x + (unsigned)it * STHREADS;
                const unsigned i = ((t & ~(j - 1)) << 1) | (t & (j - 1));
                const unsigned p = i | j;
                const float a = s[i];
                const float b = s[p];
                const bool  up = (((base + i) & k) == 0);
                const bool  sw = up ? (a < b) : (a > b);
                if (sw) { s[i] = b; s[p] = a; }
            }
            __syncthreads();
        }
        if (!FULL) break;
    }

    for (int t = threadIdx.x; t < SEGSZ / 4; t += STHREADS)
        ((float4*)g)[t] = ((const float4*)s)[t];
}

// top-k copy fused with activation: 0=none 1=PReLU(*alphap) 2=ReLU 3=ELU
__global__ void topk_act_kernel(const float* __restrict__ buf, float* __restrict__ out,
                                int n_p, int cols, int k, int mode,
                                const float* __restrict__ alphap)
{
    const size_t total = (size_t)cols * k;
    size_t idx = (size_t)blockIdx.x * blockDim.x + threadIdx.x;
    if (idx >= total) return;
    const size_t col = idx / (size_t)k;
    const int    i   = (int)(idx - col * (size_t)k);
    float v = buf[col * (size_t)n_p + i];
    if (mode == 1)      { const float a = *alphap; v = (v >= 0.f) ? v : a * v; }
    else if (mode == 2) { v = fmaxf(v, 0.f); }
    else if (mode == 3) { v = (v > 0.f) ? v : expm1f(v); }
    out[idx] = v;
}

// x: [S,16] row-major  ->  dst: [16][S]
__global__ void transpose_x_kernel(const float* __restrict__ x, float* __restrict__ dst,
                                   int S, int D)
{
    const size_t total = (size_t)S * D;
    size_t idx = (size_t)blockIdx.x * blockDim.x + threadIdx.x;
    if (idx >= total) return;
    const int w = (int)(idx / (size_t)S);
    const int h = (int)(idx % (size_t)S);
    dst[idx] = x[(size_t)h * D + w];
}

// ---------------------------------------------------------------------------
// FC: out[c][o] = bfc[o] + sum_{j<1600} act[c][j&1][j>>1] * Wfc[o][j]
// act layout [5][2][800]; j = h*2 + w.  M=5 (pad 16), N=200, K=1600.
// ---------------------------------------------------------------------------
__global__ __launch_bounds__(32)
void fc_wmma_kernel(const float* __restrict__ act, const float* __restrict__ Wfc,
                    const float* __restrict__ bfc, float* __restrict__ outfc)
{
    const int lane = threadIdx.x;
    const int m    = lane & 15;
    const int half = lane >> 4;
    const int n0   = blockIdx.x * 16;
    v8f acc = {};

#if HAS_WMMA_F32F32
    for (int kc = 0; kc < 1600; kc += 4) {
        v2f av, bv;
#pragma unroll
        for (int v = 0; v < 2; ++v) {
            const int kl = kc + 2 * half + v;
            av[v] = (m < 5) ? act[((size_t)m * 2 + (kl & 1)) * 800 + (kl >> 1)] : 0.f;
            const int n = n0 + m;
            bv[v] = (n < 200) ? Wfc[(size_t)n * 1600 + kl] : 0.f;
        }
        acc = __builtin_amdgcn_wmma_f32_16x16x4_f32(
                  false, av, false, bv, (short)0, acc, false, false);
    }
#else
    for (int kc = 0; kc < 1600; kc += 32) {
        v16h av, bv;
#pragma unroll
        for (int v = 0; v < 8; ++v) {
#pragma unroll
            for (int p = 0; p < 2; ++p) {
                const int kla = kc + 16 * (v >> 2) + 8 * half + 2 * (v & 3) + p;
                float a = (m < 5) ? act[((size_t)m * 2 + (kla & 1)) * 800 + (kla >> 1)] : 0.f;
                const int klb = kc + 16 * (v >> 2) + 8 * half + 2 * (v & 3) + p;
                const int n   = n0 + m;
                float b = (n < 200) ? Wfc[(size_t)n * 1600 + klb] : 0.f;
                av[2 * v + p] = (_Float16)a;
                bv[2 * v + p] = (_Float16)b;
            }
        }
        acc = __builtin_amdgcn_wmma_f32_16x16x32_f16(
                  false, av, false, bv, (short)0, acc, false, false);
    }
#endif

#pragma unroll
    for (int r = 0; r < 8; ++r) {
        const int mm = r + 8 * half;
        const int n  = n0 + m;
        if (mm < 5 && n < 200)
            outfc[mm * 200 + n] = acc[r] + bfc[n];
    }
}

__global__ void logsoftmax_kernel(const float* __restrict__ in, float* __restrict__ out)
{
    __shared__ float red[1024];
    const int t = threadIdx.x;
    const float v = (t < 1000) ? in[t] : -__builtin_inff();
    red[t] = v; __syncthreads();
    for (int s = 512; s > 0; s >>= 1) {
        if (t < s) red[t] = fmaxf(red[t], red[t + s]);
        __syncthreads();
    }
    const float mx = red[0]; __syncthreads();
    const float e = (t < 1000) ? expf(v - mx) : 0.f;
    red[t] = e; __syncthreads();
    for (int s = 512; s > 0; s >>= 1) {
        if (t < s) red[t] += red[t + s];
        __syncthreads();
    }
    const float lse = logf(red[0]) + mx;
    if (t < 1000) out[t] = v - lse;
}

// ---------------------------------------------------------------------------
// Host orchestration
// ---------------------------------------------------------------------------
extern "C" void kernel_launch(void* const* d_in, const int* in_sizes, int n_in,
                              void* d_out, int out_size, void* d_ws, size_t ws_size,
                              hipStream_t stream)
{
    (void)in_sizes; (void)n_in; (void)out_size; (void)ws_size;

    const float* x   = (const float*)d_in[0];
    const float* w0  = (const float*)d_in[1];
    const float* b0  = (const float*)d_in[2];
    const float* wm  = (const float*)d_in[3];   // [17][2][2][9]
    const float* bm  = (const float*)d_in[4];   // [17][2]
    const float* w18 = (const float*)d_in[5];
    const float* b18 = (const float*)d_in[6];
    const float* w19 = (const float*)d_in[7];
    const float* b19 = (const float*)d_in[8];
    const float* a1  = (const float*)d_in[9];
    const float* a2  = (const float*)d_in[10];
    const float* Wfc = (const float*)d_in[11];
    const float* bfc = (const float*)d_in[12];
    float*       out = (float*)d_out;

    float* ws  = (float*)d_ws;
    float* A   = ws;                           // 8.5M floats (34 MB)
    float* B   = ws + (size_t)8500000;         // 8.5M floats (34 MB)
    float* SB  = ws + (size_t)17000000;        // 17M  floats (68 MB) sort buffer
    float* FCB = ws + (size_t)34000000;        // FC scratch (1000 floats)

    const int S = 262144, D = 16, L = 20, KTOP = 800;

    auto launch_conv = [&](const float* in, const float* wgt, const float* bias,
                           float* o, int Cin, int Cout, int W, int Hin,
                           int Ksz, int pad) -> int {
        const int Hout  = Hin + 2 * pad - Ksz + 1;
        const int tiles = (Hout + 15) >> 4;
        conv_wmma_kernel<<<(unsigned)(W * tiles), 32, 0, stream>>>(
            in, wgt, bias, o, Cin, Cout, W, Hin, Hout, Ksz, pad);
        return Hout;
    };
    auto launch_fold = [&](const float* in, float* o, int C, int W, int H) {
        const size_t total = (size_t)C * (W / 2) * (size_t)C * H;
        fold_kernel<<<(unsigned)((total + 255) / 256), 256, 0, stream>>>(in, o, C, W, H);
    };
    auto launch_sort_topk = [&](const float* src, float* dst, int cols, int H,
                                int k, int mode, const float* alphap) {
        int n_p = 1; while (n_p < H) n_p <<= 1;
        const size_t   total  = (size_t)cols * n_p;
        const unsigned blocks = (unsigned)((total + 255) / 256);
        sort_pad_kernel<<<blocks, 256, 0, stream>>>(src, SB, n_p, cols, H);

        if (n_p >= SEGSZ) {
            const unsigned lblocks = (unsigned)((size_t)cols * (n_p / SEGSZ));
            // all stages k=2..SEGSZ, fully in LDS
            bitonic_local_kernel<true><<<lblocks, STHREADS, 0, stream>>>(SB, n_p, 0u);
            for (unsigned kk = (unsigned)SEGSZ * 2; kk <= (unsigned)n_p; kk <<= 1) {
                // cross-segment stages j >= SEGSZ in global memory
                for (unsigned j = kk >> 1; j >= (unsigned)SEGSZ; j >>= 1)
                    bitonic_step_kernel<<<blocks, 256, 0, stream>>>(SB, n_p, cols, j, kk);
                // remaining stages j < SEGSZ in LDS
                bitonic_local_kernel<false><<<lblocks, STHREADS, 0, stream>>>(SB, n_p, kk);
            }
        } else {
            for (unsigned kk = 2; kk <= (unsigned)n_p; kk <<= 1)
                for (unsigned j = kk >> 1; j > 0; j >>= 1)
                    bitonic_step_kernel<<<blocks, 256, 0, stream>>>(SB, n_p, cols, j, kk);
        }

        const size_t t2 = (size_t)cols * k;
        topk_act_kernel<<<(unsigned)((t2 + 255) / 256), 256, 0, stream>>>(
            SB, dst, n_p, cols, k, mode, alphap);
    };

    // --- input transpose: x [S,16] -> A [16][S] -------------------------------
    {
        const size_t total = (size_t)S * D;
        transpose_x_kernel<<<(unsigned)((total + 255) / 256), 256, 0, stream>>>(x, A, S, D);
    }

    // --- layer 0: conv(99,p98) -> fold -> kmax(S) -> PReLU(a1) ----------------
    int H = launch_conv(A, w0, b0, B, 1, 2, 16, S, 99, 98);  // B: [2][16][S+98]
    launch_fold(B, A, 2, 16, H);                              // A: [2][8][2H]
    H *= 2;
    launch_sort_topk(A, B, 16, H, S, /*PReLU*/1, a1);         // B: [2][8][S]
    H = S;

    // --- layers 1..17: conv(9,p8) -> kmax -> PReLU(a2)/ReLU -------------------
    for (int i = 0; i < 17; ++i) {
        int k = (int)(((double)(L - 1 - i) / (double)L) * (double)S);
        if (k < KTOP) k = KTOP;
        const int Hc = launch_conv(B, wm + (size_t)i * 36, bm + (size_t)i * 2,
                                   A, 2, 2, 8, H, 9, 8);      // A: [2][8][Hc]
        launch_sort_topk(A, B, 16, Hc, k, (i == 0) ? 1 : 2, a2);
        H = k;                                                // B: [2][8][k]
    }

    // --- layer 18: conv(9,p8) -> fold -> kmax -> ELU --------------------------
    {
        int k18 = (int)(((double)(L - 18) / (double)L) * (double)S);
        if (k18 < KTOP) k18 = KTOP;
        const int Hc = launch_conv(B, w18, b18, A, 2, 2, 8, H, 9, 8); // A [2][8][Hc]
        launch_fold(A, B, 2, 8, Hc);                                  // B [2][4][2Hc]
        launch_sort_topk(B, A, 8, 2 * Hc, k18, /*ELU*/3, nullptr);    // A [2][4][k18]
        H = k18;
    }

    // --- layer 19: conv(358,p357) -> fold -> kmax(800) -> ELU -----------------
    {
        const int Hc = launch_conv(A, w19, b19, B, 2, 5, 4, H, 358, 357); // B [5][4][Hc]
        launch_fold(B, A, 5, 4, Hc);                                      // A [5][2][5Hc]
        launch_sort_topk(A, B, 10, 5 * Hc, KTOP, /*ELU*/3, nullptr);      // B [5][2][800]
    }

    // --- FC (WMMA, 5x1600x200) + log_softmax over 1000 ------------------------
    fc_wmma_kernel<<<13, 32, 0, stream>>>(B, Wfc, bfc, FCB);
    logsoftmax_kernel<<<1, 1024, 0, stream>>>(FCB, out);
}
